// Block_67284957659365
// MI455X (gfx1250) — matrix operand order repaired and checked by reading, not compiled, not run on hardware
//
#include <hip/hip_runtime.h>
#include <hip/hip_bf16.h>

typedef __bf16 bf16_t;
typedef __attribute__((ext_vector_type(16))) __bf16 v16bf;
typedef __attribute__((ext_vector_type(8)))  float  v8f;

#define TOK   8192      // 8 * 1024 tokens
#define DIM   768
#define DIM3  2304
#define HID   3072
#define NHEAD 12
#define DH    64

__device__ inline v8f v8f_zero() {
    v8f v;
#pragma unroll
    for (int i = 0; i < 8; ++i) v[i] = 0.0f;
    return v;
}

__device__ inline v8f wmma_bf16(v16bf a, v16bf b, v8f c) {
    // D = A(16x32 bf16) x B(32x16 bf16) + C(16x16 f32)
    return __builtin_amdgcn_wmma_f32_16x16x32_bf16(
        /*neg_a=*/false, a, /*neg_b=*/false, b,
        /*c_mod=*/(short)0, c, /*reuse_a=*/false, /*reuse_b=*/false);
}

// A-fragment: 16(M) x 32(K) bf16, row-major source with leading dim ld.
// Lane L: m = L&15, hi = L>>4.  VGPR p: K = 2p+16hi.
__device__ inline v16bf load_a_frag(const bf16_t* a, int ld, int k0) {
    int lane = threadIdx.x & 31;
    int m = lane & 15, hi = lane >> 4;
    const bf16_t* p = a + (size_t)m * ld + k0;
    v16bf r;
#pragma unroll
    for (int pr = 0; pr < 8; ++pr) {
        int k = 2 * pr + 16 * hi;
        r[2 * pr]     = p[k];
        r[2 * pr + 1] = p[k + 1];
    }
    return r;
}

// B-fragment: 32(K) x 16(N) bf16 read from a K-transposed [N][K] buffer.
// Lane L: n = L&15, hi = L>>4, element e -> k = k0 + 16*hi + e.
__device__ inline v16bf load_b_frag(const bf16_t* bt, int ld, int k0) {
    int lane = threadIdx.x & 31;
    int n = lane & 15, hi = lane >> 4;
    const bf16_t* p = bt + (size_t)n * ld + k0 + 16 * hi;
    v16bf b;
#pragma unroll
    for (int e = 0; e < 16; ++e) b[e] = p[e];
    return b;
}

// ---------------------------------------------------------------------------
// Weight convert + transpose: W f32 [K][N] -> Wt bf16 [N][K]
// ---------------------------------------------------------------------------
__global__ __launch_bounds__(256) void conv_transpose_kernel(
        const float* __restrict__ W, bf16_t* __restrict__ Wt, int K, int N) {
    int idx = blockIdx.x * 256 + threadIdx.x;   // idx = n*K + k
    int total = K * N;
    if (idx >= total) return;
    int n = idx / K;
    int k = idx - n * K;
    Wt[idx] = (bf16_t)W[(size_t)k * N + n];
}

// ---------------------------------------------------------------------------
// LayerNorm: one wave per token (C = 768 -> 24 elements/lane), bf16 output.
// ---------------------------------------------------------------------------
__global__ __launch_bounds__(256) void ln_kernel(
        const float* __restrict__ x, const float* __restrict__ g,
        const float* __restrict__ b, bf16_t* __restrict__ out) {
    int tok  = blockIdx.x * 8 + (threadIdx.x >> 5);
    int lane = threadIdx.x & 31;
    const float* row = x + (size_t)tok * DIM;
    float vals[24];
    float s = 0.0f;
#pragma unroll
    for (int i = 0; i < 24; ++i) { vals[i] = row[lane + i * 32]; s += vals[i]; }
#pragma unroll
    for (int m = 16; m >= 1; m >>= 1) s += __shfl_xor(s, m, 32);
    float mu = s * (1.0f / DIM);
    float v = 0.0f;
#pragma unroll
    for (int i = 0; i < 24; ++i) { float d = vals[i] - mu; v += d * d; }
#pragma unroll
    for (int m = 16; m >= 1; m >>= 1) v += __shfl_xor(v, m, 32);
    float rstd = rsqrtf(v * (1.0f / DIM) + 1e-5f);
    bf16_t* orow = out + (size_t)tok * DIM;
#pragma unroll
    for (int i = 0; i < 24; ++i) {
        int c = lane + i * 32;
        orow[c] = (bf16_t)((vals[i] - mu) * rstd * g[c] + b[c]);
    }
}

// ---------------------------------------------------------------------------
// WMMA GEMM: out[M,N] = A[M,K](bf16) x Wt[N,K](bf16)^T + bias, epilogue MODE:
//   0: f32 out   1: bf16 out   2: GELU -> bf16 out   3: residual + f32 out
// Block tile 128x128, 256 threads = 8 waves (4 in M x 2 in N),
// wave tile 32x64 -> 8 WMMA per 32-deep k-step.
// ---------------------------------------------------------------------------
template <int MODE>
__global__ __launch_bounds__(256) void gemm_kernel(
        const bf16_t* __restrict__ A, const bf16_t* __restrict__ Bt,
        const float* __restrict__ bias, const float* __restrict__ residual,
        void* __restrict__ outp, int M, int N, int K) {
    __shared__ bf16_t As[128 * 32];
    __shared__ bf16_t Bs[128 * 32];

    int m0 = blockIdx.y * 128;
    int n0 = blockIdx.x * 128;
    int t = threadIdx.x;
    int wid = t >> 5, lane = t & 31;
    int wm = wid & 3;        // wave row (M): 4 waves x 32 rows
    int wn = wid >> 2;       // wave col (N): 2 waves x 64 cols

    v8f acc[2][4];
#pragma unroll
    for (int i = 0; i < 2; ++i)
#pragma unroll
        for (int j = 0; j < 4; ++j) acc[i][j] = v8f_zero();

    const int srow = t >> 1, scol = (t & 1) * 16;   // 16 bf16 per thread per tile

    for (int k0 = 0; k0 < K; k0 += 32) {
        // Stage A tile (128x32) and B tile (128 rows of Wt x 32 K)
        {
            const bf16_t* srcA = A + (size_t)(m0 + srow) * K + k0 + scol;
            const uint4* sa = reinterpret_cast<const uint4*>(srcA);
            uint4* da = reinterpret_cast<uint4*>(&As[srow * 32 + scol]);
            da[0] = sa[0];
            da[1] = sa[1];
            const bf16_t* srcB = Bt + (size_t)(n0 + srow) * K + k0 + scol;
            const uint4* sb = reinterpret_cast<const uint4*>(srcB);
            uint4* db = reinterpret_cast<uint4*>(&Bs[srow * 32 + scol]);
            db[0] = sb[0];
            db[1] = sb[1];
            if (k0 + 32 < K) {
                __builtin_prefetch(srcA + 32, 0, 1);
                __builtin_prefetch(srcB + 32, 0, 1);
            }
        }
        __syncthreads();

        v16bf a0 = load_a_frag(As + (wm * 32) * 32, 32, 0);
        v16bf a1 = load_a_frag(As + (wm * 32 + 16) * 32, 32, 0);
#pragma unroll
        for (int j = 0; j < 4; ++j) {
            v16bf bj = load_b_frag(Bs + (wn * 64 + j * 16) * 32, 32, 0);
            acc[0][j] = wmma_bf16(a0, bj, acc[0][j]);
            acc[1][j] = wmma_bf16(a1, bj, acc[1][j]);
        }
        __syncthreads();
    }

    // Epilogue. C layout: lane n = lane&15, VGPR v -> row v + 8*(lane>>4).
    int nl = lane & 15, hi = lane >> 4;
#pragma unroll
    for (int i = 0; i < 2; ++i) {
#pragma unroll
        for (int j = 0; j < 4; ++j) {
            int colbase = n0 + wn * 64 + j * 16 + nl;
            float bv = bias[colbase];
#pragma unroll
            for (int v = 0; v < 8; ++v) {
                int row = m0 + wm * 32 + i * 16 + v + 8 * hi;
                size_t idx = (size_t)row * N + colbase;
                float val = acc[i][j][v] + bv;
                if (MODE == 0) {
                    ((float*)outp)[idx] = val;
                } else if (MODE == 1) {
                    ((bf16_t*)outp)[idx] = (bf16_t)val;
                } else if (MODE == 2) {
                    float gel = 0.5f * val * (1.0f + erff(val * 0.70710678118654752f));
                    ((bf16_t*)outp)[idx] = (bf16_t)gel;
                } else {
                    ((float*)outp)[idx] = residual[idx] + val;
                }
            }
        }
    }
}

// ---------------------------------------------------------------------------
// Flash attention: grid = B*H*(N/64) blocks of 128 threads (4 waves).
// Each wave owns 16 query rows; loop over 64-key chunks (16 WMMA per online-
// softmax round: 8 for S=QK^T, 8 for O+=P.V).
// qkv layout: [token][2304] bf16, q/k/v at col 0/768/1536, head h at +h*64.
// ---------------------------------------------------------------------------
__global__ __launch_bounds__(128) void attn_kernel(
        const bf16_t* __restrict__ qkv, bf16_t* __restrict__ out) {
    const int SEQ = 1024;
    int blk  = blockIdx.x;
    int qblk = blk & 15;          // 16 query blocks of 64
    int bh   = blk >> 4;          // 0..95
    int h    = bh % NHEAD;
    int b    = bh / NHEAD;
    int lane = threadIdx.x & 31;
    int wid  = threadIdx.x >> 5;  // 0..3
    int q0   = qblk * 64 + wid * 16;

    const bf16_t* qbase = qkv + (size_t)(b * SEQ) * DIM3 + h * DH;
    const bf16_t* kbase = qbase + DIM;
    const bf16_t* vbase = qbase + 2 * DIM;

    __shared__ bf16_t Kc[64 * 64];      // [key][d]  (B-frags for S)
    __shared__ bf16_t Vt[64 * 64];      // [d][key]  (B-frags for O)
    __shared__ bf16_t Pl[4][16 * 64];   // per-wave P relayout scratch

    // Q fragments (persist across key loop)
    v16bf qa0 = load_a_frag(qbase + (size_t)q0 * DIM3, DIM3, 0);
    v16bf qa1 = load_a_frag(qbase + (size_t)q0 * DIM3, DIM3, 32);

    float rm[8], rs[8];
#pragma unroll
    for (int v = 0; v < 8; ++v) { rm[v] = -3.0e38f; rs[v] = 0.0f; }
    v8f o[4];
#pragma unroll
    for (int i = 0; i < 4; ++i) o[i] = v8f_zero();

    const float scale = 0.125f;   // 1/sqrt(64)
    int t = threadIdx.x;

    for (int kc = 0; kc < SEQ; kc += 64) {
        // Stage K chunk [64 keys][64 d]: 32 bf16 per thread
        {
            int key = t >> 1, col = (t & 1) * 32;
            const bf16_t* src = kbase + (size_t)(kc + key) * DIM3 + col;
            const uint4* s4 = reinterpret_cast<const uint4*>(src);
            uint4* d4 = reinterpret_cast<uint4*>(&Kc[key * 64 + col]);
#pragma unroll
            for (int i = 0; i < 4; ++i) d4[i] = s4[i];
        }
        // Stage V chunk transposed [64 d][64 keys]
        {
            int d = t & 63, half = t >> 6;
#pragma unroll
            for (int i = 0; i < 32; ++i)
                Vt[d * 64 + half * 32 + i] =
                    vbase[(size_t)(kc + half * 32 + i) * DIM3 + d];
        }
        __syncthreads();

        // S = Q x K^T: 16 queries x 64 keys (4 key groups)
        v8f s[4];
#pragma unroll
        for (int g = 0; g < 4; ++g) {
            v8f sg = v8f_zero();
            v16bf bk0 = load_b_frag(Kc + g * 16 * 64, 64, 0);
            sg = wmma_bf16(qa0, bk0, sg);
            v16bf bk1 = load_b_frag(Kc + g * 16 * 64, 64, 32);
            sg = wmma_bf16(qa1, bk1, sg);
            s[g] = sg;
        }

        // Online softmax (rows live in VGPR v, cols across 16 lanes x 4 groups)
        float fct[8];
        int key = lane & 15, hi = lane >> 4;
#pragma unroll
        for (int v = 0; v < 8; ++v) {
            float a0 = s[0][v] * scale, a1 = s[1][v] * scale;
            float a2 = s[2][v] * scale, a3 = s[3][v] * scale;
            float mx = fmaxf(fmaxf(a0, a1), fmaxf(a2, a3));
#pragma unroll
            for (int m = 8; m >= 1; m >>= 1) mx = fmaxf(mx, __shfl_xor(mx, m, 32));
            float nm  = fmaxf(rm[v], mx);
            float fac = __expf(rm[v] - nm);
            rm[v] = nm;
            float e0 = __expf(a0 - nm), e1 = __expf(a1 - nm);
            float e2 = __expf(a2 - nm), e3 = __expf(a3 - nm);
            float ps = (e0 + e1) + (e2 + e3);
#pragma unroll
            for (int m = 8; m >= 1; m >>= 1) ps += __shfl_xor(ps, m, 32);
            rs[v] = rs[v] * fac + ps;
            fct[v] = fac;
            int m_row = v + 8 * hi;
            bf16_t* prow = &Pl[wid][m_row * 64 + key];
            prow[0]  = (bf16_t)e0;
            prow[16] = (bf16_t)e1;
            prow[32] = (bf16_t)e2;
            prow[48] = (bf16_t)e3;
        }
#pragma unroll
        for (int nt = 0; nt < 4; ++nt)
#pragma unroll
            for (int v = 0; v < 8; ++v) o[nt][v] *= fct[v];

        // P fragments (16 x 64) from per-wave LDS scratch
        v16bf pa0 = load_a_frag(Pl[wid], 64, 0);
        v16bf pa1 = load_a_frag(Pl[wid], 64, 32);

        // O += P x V  (n = d columns, k = 64 keys in 2 steps)
#pragma unroll
        for (int nt = 0; nt < 4; ++nt) {
            v16bf bv0 = load_b_frag(Vt + nt * 16 * 64, 64, 0);
            o[nt] = wmma_bf16(pa0, bv0, o[nt]);
            v16bf bv1 = load_b_frag(Vt + nt * 16 * 64, 64, 32);
            o[nt] = wmma_bf16(pa1, bv1, o[nt]);
        }
        __syncthreads();
    }

    // Normalize and write bf16 output [token][768]
    int nl = lane & 15, hi = lane >> 4;
#pragma unroll
    for (int nt = 0; nt < 4; ++nt) {
#pragma unroll
        for (int v = 0; v < 8; ++v) {
            int m = v + 8 * hi;
            int tokrow = b * SEQ + q0 + m;
            int col = h * DH + nt * 16 + nl;
            out[(size_t)tokrow * DIM + col] = (bf16_t)(o[nt][v] / rs[v]);
        }
    }
}

// ---------------------------------------------------------------------------
extern "C" void kernel_launch(void* const* d_in, const int* in_sizes, int n_in,
                              void* d_out, int out_size, void* d_ws, size_t ws_size,
                              hipStream_t stream) {
    const float* x      = (const float*)d_in[0];
    const float* ln1_g  = (const float*)d_in[1];
    const float* ln1_b  = (const float*)d_in[2];
    const float* qkv_w  = (const float*)d_in[3];
    const float* qkv_b  = (const float*)d_in[4];
    const float* proj_w = (const float*)d_in[5];
    const float* proj_b = (const float*)d_in[6];
    const float* ln2_g  = (const float*)d_in[7];
    const float* ln2_b  = (const float*)d_in[8];
    const float* fc1_w  = (const float*)d_in[9];
    const float* fc1_b  = (const float*)d_in[10];
    const float* fc2_w  = (const float*)d_in[11];
    const float* fc2_b  = (const float*)d_in[12];
    float* out = (float*)d_out;

    char* base = (char*)d_ws;
    size_t off = 0;
    auto take = [&](size_t bytes) {
        char* p = base + off;
        off = (off + bytes + 255) & ~(size_t)255;
        return p;
    };
    bf16_t* wt_qkv = (bf16_t*)take((size_t)DIM * DIM3 * 2);
    bf16_t* wt_prj = (bf16_t*)take((size_t)DIM * DIM * 2);
    bf16_t* wt_fc1 = (bf16_t*)take((size_t)DIM * HID * 2);
    bf16_t* wt_fc2 = (bf16_t*)take((size_t)HID * DIM * 2);
    bf16_t* h1     = (bf16_t*)take((size_t)TOK * DIM * 2);
    bf16_t* qkvbuf = (bf16_t*)take((size_t)TOK * DIM3 * 2);
    bf16_t* attn_o = (bf16_t*)take((size_t)TOK * DIM * 2);
    float*  x2     = (float*) take((size_t)TOK * DIM * 4);
    bf16_t* h2     = (bf16_t*)take((size_t)TOK * DIM * 2);
    bf16_t* fc1o   = (bf16_t*)take((size_t)TOK * HID * 2);

    // 1) weight convert + transpose (f32 [K,N] -> bf16 [N,K])
    conv_transpose_kernel<<<(DIM * DIM3 + 255) / 256, 256, 0, stream>>>(qkv_w, wt_qkv, DIM, DIM3);
    conv_transpose_kernel<<<(DIM * DIM  + 255) / 256, 256, 0, stream>>>(proj_w, wt_prj, DIM, DIM);
    conv_transpose_kernel<<<(DIM * HID  + 255) / 256, 256, 0, stream>>>(fc1_w, wt_fc1, DIM, HID);
    conv_transpose_kernel<<<(HID * DIM  + 255) / 256, 256, 0, stream>>>(fc2_w, wt_fc2, HID, DIM);

    // 2) LN1 -> h1 (bf16)
    ln_kernel<<<TOK / 8, 256, 0, stream>>>(x, ln1_g, ln1_b, h1);

    // 3) QKV GEMM: [8192,768] x [768,2304] -> bf16
    gemm_kernel<1><<<dim3(DIM3 / 128, TOK / 128), 256, 0, stream>>>(
        h1, wt_qkv, qkv_b, nullptr, qkvbuf, TOK, DIM3, DIM);

    // 4) Flash attention -> attn_o (bf16)
    attn_kernel<<<16 * 8 * NHEAD, 128, 0, stream>>>(qkvbuf, attn_o);

    // 5) proj GEMM + residual: x2 = x + attn_o @ proj_w + b  (f32)
    gemm_kernel<3><<<dim3(DIM / 128, TOK / 128), 256, 0, stream>>>(
        attn_o, wt_prj, proj_b, x, x2, TOK, DIM, DIM);

    // 6) LN2 -> h2 (bf16)
    ln_kernel<<<TOK / 8, 256, 0, stream>>>(x2, ln2_g, ln2_b, h2);

    // 7) FC1 GEMM + exact GELU -> bf16
    gemm_kernel<2><<<dim3(HID / 128, TOK / 128), 256, 0, stream>>>(
        h2, wt_fc1, fc1_b, nullptr, fc1o, TOK, HID, DIM);

    // 8) FC2 GEMM + residual -> d_out (f32)
    gemm_kernel<3><<<dim3(DIM / 128, TOK / 128), 256, 0, stream>>>(
        fc1o, wt_fc2, fc2_b, x2, out, TOK, DIM, HID);
}
